// Net_59459527245898
// MI455X (gfx1250) — matrix operand order repaired and checked by reading, not compiled, not run on hardware
//
#include <hip/hip_runtime.h>
#include <math.h>

#define NN 8
#define CC 21
#define HW 4096
#define CF 768
#define JC 48      // padded center rows (21*2 = 42 -> 48)
#define JA 42

typedef __attribute__((ext_vector_type(16))) __bf16 v16bf;
typedef __attribute__((ext_vector_type(8)))  __bf16 v8bf;
typedef __attribute__((ext_vector_type(8)))  float  v8f;

__device__ inline __bf16 f2bf(float f) {
  union { float f; unsigned u; } x; x.f = f;
  unsigned r = (x.u + 0x7FFFu + ((x.u >> 16) & 1u)) >> 16;
  union { unsigned short s; __bf16 b; } y; y.s = (unsigned short)r;
  return y.b;
}

// ---------------- utility fills ----------------
__global__ void fill_f32(float* p, float v, int n) {
  int i = blockIdx.x * blockDim.x + threadIdx.x;
  if (i < n) p[i] = v;
}
__global__ void fill_u32(unsigned* p, unsigned v, int n) {
  int i = blockIdx.x * blockDim.x + threadIdx.x;
  if (i < n) p[i] = v;
}
__global__ void fill_i32(int* p, int v, int n) {
  int i = blockIdx.x * blockDim.x + threadIdx.x;
  if (i < n) p[i] = v;
}

// ---------------- seeds: per-pixel argmax class, gated by label ----------------
__global__ void k_seed(const float* __restrict__ cam, const float* __restrict__ label,
                       int* __restrict__ belong, int* __restrict__ cnt) {
  int e = blockIdx.x * blockDim.x + threadIdx.x;   // NN*HW
  if (e >= NN * HW) return;
  int n = e >> 12, m = e & 4095;
  const float* p = cam + (size_t)n * CC * HW + m;
  float best = p[0]; int bc = 0;
  for (int c = 1; c < CC; ++c) {
    float v = p[(size_t)c * HW];
    if (v > best) { best = v; bc = c; }            // strict > == first-max (jnp.argmax)
  }
  int valid = label[n * CC + bc] > 0.f;
  belong[e] = valid ? bc : -1;
  if (valid) atomicAdd(&cnt[n * CC + bc], 1);
}

// ---------------- per-pixel feature norms ----------------
__global__ void k_norm(const float* __restrict__ feat, float* __restrict__ invn,
                       float* __restrict__ rowsc) {
  int e = blockIdx.x * blockDim.x + threadIdx.x;   // NN*HW
  if (e >= NN * HW) return;
  int n = e >> 12, m = e & 4095;
  const float* p = feat + (size_t)n * CF * HW + m;
  float s = 0.f;
  for (int d = 0; d < CF; ++d) { float v = p[(size_t)d * HW]; s += v * v; }
  float nf = sqrtf(s);
  float cl = fmaxf(nf, 1e-8f);
  invn[e] = 1.f / cl;                  // for feat_n = feat / max(||f||,1e-8)
  rowsc[e] = cl / (nf + 1e-5f);        // feat_ot = feat_n * rowsc
}

// ---------------- transpose [n,cf,hw] -> normalized bf16 [n,hw,cf] ----------------
__global__ void __launch_bounds__(256)
k_transpose(const float* __restrict__ feat, const float* __restrict__ invn,
            __bf16* __restrict__ featn) {
  __shared__ float tile[32][33];
  int m0 = blockIdx.x * 32, d0 = blockIdx.y * 32, n = blockIdx.z;
  int tx = threadIdx.x, ty = threadIdx.y;
#pragma unroll
  for (int s = 0; s < 4; ++s) {
    int d = d0 + ty + s * 8;
    tile[ty + s * 8][tx] = feat[((size_t)n * CF + d) * HW + m0 + tx];
  }
  __syncthreads();
#pragma unroll
  for (int s = 0; s < 4; ++s) {
    int m = m0 + ty + s * 8;
    float v = tile[tx][ty + s * 8] * invn[n * HW + m];
    featn[((size_t)n * HW + m) * CF + d0 + tx] = f2bf(v);
  }
}

// ---------------- k-means init: first two valid pixels per (n,c) ----------------
__global__ void k_scan(const int* __restrict__ belong, int* __restrict__ idx0,
                       int* __restrict__ idx1) {
  int e = blockIdx.x * blockDim.x + threadIdx.x;   // NN*CC
  if (e >= NN * CC) return;
  int n = e / CC, c = e % CC;
  const int* b = belong + n * HW;
  int i0 = -1, i1 = -1;
  for (int m = 0; m < HW; ++m) {
    if (b[m] == c) {
      if (i0 < 0) i0 = m;
      else { i1 = m; break; }
    }
  }
  idx0[e] = i0; idx1[e] = i1;
}

__global__ void k_initc(const float* __restrict__ feat, const int* __restrict__ idx0,
                        const int* __restrict__ idx1, float* __restrict__ centers) {
  int blk = blockIdx.x;                 // NN*JC
  int n = blk / JC, j = blk % JC;
  float* out = centers + ((size_t)n * JC + j) * CF;
  int src = -1;
  if (j < JA) {
    int e = n * CC + (j >> 1);
    src = (j & 1) ? idx1[e] : idx0[e];
  }
  for (int d = threadIdx.x; d < CF; d += blockDim.x) {
    float v = 0.f;
    if (src >= 0) v = feat[((size_t)n * CF + d) * HW + src];
    out[d] = v;
  }
}

// ---------------- center normalization -> bf16 (+ Sinkhorn column rescale) ----------------
__global__ void __launch_bounds__(256)
k_cnorm(const float* __restrict__ centers, __bf16* __restrict__ ctrn,
        float* __restrict__ colsc) {
  int blk = blockIdx.x;                 // NN*JC
  int n = blk / JC, j = blk % JC;
  const float* in = centers + ((size_t)n * JC + j) * CF;
  __shared__ float red[256];
  float s = 0.f;
  for (int d = threadIdx.x; d < CF; d += 256) { float v = in[d]; s += v * v; }
  red[threadIdx.x] = s; __syncthreads();
  for (int w = 128; w > 0; w >>= 1) {
    if (threadIdx.x < w) red[threadIdx.x] += red[threadIdx.x + w];
    __syncthreads();
  }
  float nf = sqrtf(red[0]);
  float cl = fmaxf(nf, 1e-8f);
  float inv = 1.f / cl;
  __bf16* out = ctrn + ((size_t)n * JC + j) * CF;
  for (int d = threadIdx.x; d < CF; d += 256) out[d] = f2bf(in[d] * inv);
  if (threadIdx.x == 0) colsc[n * JC + j] = cl / (nf + 1e-5f);
}

// ---------------- WMMA GEMM: sim[n,4096,48] = feat_n @ centers_nT ----------------
__global__ void __launch_bounds__(256)
k_gemm(const __bf16* __restrict__ featn, const __bf16* __restrict__ ctrn,
       float* __restrict__ sim) {
  int wave = threadIdx.x >> 5, lane = threadIdx.x & 31;
  int tile = blockIdx.x * 8 + wave;     // 2048 tiles: 256 per image
  int n = tile >> 8, mt = tile & 255;
  int m0 = mt << 4;
  int half = lane >> 4, l16 = lane & 15;
  const __bf16* A = featn + (size_t)n * HW * CF;
  const __bf16* B = ctrn + (size_t)n * JC * CF;
  const __bf16* arow = A + (size_t)(m0 + l16) * CF;
  v8f acc0 = {}, acc1 = {}, acc2 = {};
  for (int k0 = 0; k0 < CF; k0 += 32) {
    // A fragment: row l16, K = k0 + half*16 + {0..15} (contiguous 32B)
    const __bf16* ap = arow + k0 + half * 16;
    v16bf a = *(const v16bf*)(ap);
    // B fragments: column j0+l16, K = k0 + half*16 + {0..15} (contiguous 32B)
    int kb = k0 + half * 16;
    v16bf b0 = *(const v16bf*)(B + (size_t)(0 + l16) * CF + kb);
    v16bf b1 = *(const v16bf*)(B + (size_t)(16 + l16) * CF + kb);
    v16bf b2 = *(const v16bf*)(B + (size_t)(32 + l16) * CF + kb);
    acc0 = __builtin_amdgcn_wmma_f32_16x16x32_bf16(false, a, false, b0, (short)0, acc0, false, false);
    acc1 = __builtin_amdgcn_wmma_f32_16x16x32_bf16(false, a, false, b1, (short)0, acc1, false, false);
    acc2 = __builtin_amdgcn_wmma_f32_16x16x32_bf16(false, a, false, b2, (short)0, acc2, false, false);
  }
#pragma unroll
  for (int r = 0; r < 8; ++r) {
    int m = m0 + r + 8 * half;
    float* srow = sim + ((size_t)n * HW + m) * JC;
    srow[0 + l16]  = acc0[r];
    srow[16 + l16] = acc1[r];
    srow[32 + l16] = acc2[r];
  }
}

// ---------------- assignment + masked center accumulation ----------------
__global__ void __launch_bounds__(256)
k_assign(const float* __restrict__ feat, const int* __restrict__ belong,
         const float* __restrict__ sim, float* __restrict__ cacc,
         unsigned* __restrict__ ccnt) {
  int grp = threadIdx.x >> 5, lane = threadIdx.x & 31;
  int pix = blockIdx.x * 8 + grp;       // NN*HW/8 blocks
  int n = pix >> 12, m = pix & 4095;
  int cls = belong[pix];
  if (cls < 0) return;
  const float* sr = sim + (size_t)pix * JC;
  float s0 = sr[2 * cls], s1 = sr[2 * cls + 1];
  int j = 2 * cls + ((s1 > s0) ? 1 : 0);
  float* out = cacc + ((size_t)n * JC + j) * CF;
  const float* f = feat + (size_t)n * CF * HW + m;
  for (int d = lane; d < CF; d += 32) atomicAdd(&out[d], f[(size_t)d * HW]);
  if (lane == 0) atomicAdd(&ccnt[n * JC + j], 1u);
}

__global__ void k_div(const float* __restrict__ cacc, const unsigned* __restrict__ ccnt,
                      float* __restrict__ centers) {
  int e = blockIdx.x * blockDim.x + threadIdx.x;   // NN*JC*CF
  if (e >= NN * JC * CF) return;
  unsigned c = ccnt[e / CF];
  centers[e] = cacc[e] / (float)(c ? c : 1u);
}

// ---------------- zero centers of classes with < 2 seed pixels ----------------
__global__ void k_gate(const int* __restrict__ cnt, float* __restrict__ centers) {
  int e = blockIdx.x * blockDim.x + threadIdx.x;   // NN*JC*CF
  if (e >= NN * JC * CF) return;
  int nj = e / CF, n = nj / JC, j = nj % JC;
  bool z = (j >= JA) || (cnt[n * CC + (j >> 1)] < 2);
  if (z) centers[e] = 0.f;
}

// ---------------- cam + label gate + Sinkhorn kernel matrix ----------------
__global__ void k_campost(const float* __restrict__ sim, const int* __restrict__ cnt,
                          const float* __restrict__ label, const float* __restrict__ rowsc,
                          const float* __restrict__ colsc, float* __restrict__ out1,
                          float* __restrict__ Kmat) {
  int e = blockIdx.x * blockDim.x + threadIdx.x;   // NN*CC*HW; e = (n*CC+c)*HW + m
  if (e >= NN * CC * HW) return;
  int m = e & 4095;
  int nc = e >> 12;
  int c = nc % CC, n = nc / CC;
  const float* sr = sim + ((size_t)n * HW + m) * JC;
  float camv = 0.f;
  if (cnt[nc] >= 2) camv = 0.5f * (sr[2 * c] + sr[2 * c + 1]);
  out1[e] = camv * label[nc];
  float rs = rowsc[n * HW + m];
#pragma unroll
  for (int k = 0; k < 2; ++k) {
    int j = k * CC + c;                 // faithful to reshape(n,hw,K,c).permute
    float s = sr[j] * rs * colsc[n * JC + j];
    Kmat[(size_t)e * 2 + k] = expf(-(1.f - s) * 10.f);   // 1/EPS_OT
  }
}

__global__ void __launch_bounds__(256)
k_max(const float* __restrict__ out1, float* __restrict__ maxv) {
  int nc = blockIdx.x;                  // NN*CC
  __shared__ float red[256];
  const float* p = out1 + (size_t)nc * HW;
  float mv = -INFINITY;
  for (int i = threadIdx.x; i < HW; i += 256) mv = fmaxf(mv, p[i]);
  red[threadIdx.x] = mv; __syncthreads();
  for (int w = 128; w > 0; w >>= 1) {
    if (threadIdx.x < w) red[threadIdx.x] = fmaxf(red[threadIdx.x], red[threadIdx.x + w]);
    __syncthreads();
  }
  if (threadIdx.x == 0) maxv[nc] = red[0];
}

__global__ void k_normout(float* __restrict__ out1, const float* __restrict__ maxv) {
  int e = blockIdx.x * blockDim.x + threadIdx.x;
  if (e >= NN * CC * HW) return;
  out1[e] /= (maxv[e >> 12] + 1e-5f);
}

// ---------------- Sinkhorn: one 1024-thread WG per image, L2-resident ----------------
#define SINK_ITEMS 84                    // 21*4096 / 1024
__global__ void __launch_bounds__(1024)
k_sinkhorn(const float* __restrict__ Kmat, float* __restrict__ rbuf,
           float* __restrict__ Tout) {
  int n = blockIdx.x;
  int t = threadIdx.x;
  __shared__ float s_sh[JA];
  __shared__ float cc_sh[JA];
  __shared__ float err_sh;
  const float* Kn = Kmat + (size_t)n * CC * HW * 2;
  float* rn_ = rbuf + (size_t)n * CC * HW;
  if (t < JA) cc_sh[t] = 1.f;
  for (int i = 0; i < SINK_ITEMS; ++i) rn_[t * SINK_ITEMS + i] = 1.f;  // thread-owned
  __syncthreads();
  const float u = 1.f / (float)HW;
  float err = INFINITY;
  int it = 0;
  while (it < 100 && err >= 0.01f) {
    if (t < JA) s_sh[t] = 0.f;
    if (t == 0) err_sh = 0.f;
    __syncthreads();
    float ea = 0.f;
    for (int i = 0; i < SINK_ITEMS; ++i) {
      int e = t * SINK_ITEMS + i;
      int cidx = e >> 12;
      float K0 = Kn[(size_t)e * 2 + 0];
      float K1 = Kn[(size_t)e * 2 + 1];
      float denom = K0 * cc_sh[2 * cidx] + K1 * cc_sh[2 * cidx + 1];
      float rv = u / denom;
      atomicAdd(&s_sh[2 * cidx + 0], K0 * rv);
      atomicAdd(&s_sh[2 * cidx + 1], K1 * rv);
      ea += fabsf(rv - rn_[e]);
      rn_[e] = rv;
    }
    atomicAdd(&err_sh, ea);
    __syncthreads();
    if (t < JA) cc_sh[t] = 0.5f / s_sh[t];          // v = 1/NUM_CLUSTERS
    if (t == 0) err_sh *= (1.f / (float)(CC * HW));
    __syncthreads();
    err = err_sh;
    __syncthreads();                                 // all read err before next zero
    ++it;
  }
  // T[n,c,k,h,w] = r * c * K
  for (int i = 0; i < SINK_ITEMS; ++i) {
    int e = t * SINK_ITEMS + i;
    int cidx = e >> 12, m = e & 4095;
    float rv = rn_[e];
#pragma unroll
    for (int k = 0; k < 2; ++k) {
      Tout[(((size_t)n * CC + cidx) * 2 + k) * HW + m] =
          rv * cc_sh[2 * cidx + k] * Kn[(size_t)e * 2 + k];
    }
  }
}

extern "C" void kernel_launch(void* const* d_in, const int* in_sizes, int n_in,
                              void* d_out, int out_size, void* d_ws, size_t ws_size,
                              hipStream_t stream) {
  (void)in_sizes; (void)n_in; (void)out_size; (void)ws_size;
  const float* norm_cam = (const float*)d_in[0];
  const float* label    = (const float*)d_in[1];
  const float* feature  = (const float*)d_in[2];
  float* out1 = (float*)d_out;                       // [8,21,64,64]
  float* Tout = out1 + (size_t)NN * CC * HW;         // [8,21,2,64,64]

  char* w = (char*)d_ws; size_t o = 0;
  auto take = [&](size_t bytes) -> void* {
    void* p = w + o; o = (o + bytes + 255) & ~(size_t)255; return p;
  };
  __bf16*   featn   = (__bf16*)  take((size_t)NN * HW * CF * 2);
  __bf16*   ctrn    = (__bf16*)  take((size_t)NN * JC * CF * 2);
  float*    centers = (float*)   take((size_t)NN * JC * CF * 4);
  float*    cacc    = (float*)   take((size_t)NN * JC * CF * 4);
  unsigned* ccnt    = (unsigned*)take((size_t)NN * JC * 4);
  float*    invn    = (float*)   take((size_t)NN * HW * 4);
  float*    rowsc   = (float*)   take((size_t)NN * HW * 4);
  int*      belong  = (int*)     take((size_t)NN * HW * 4);
  int*      cnt     = (int*)     take((size_t)NN * CC * 4);
  int*      idx0    = (int*)     take((size_t)NN * CC * 4);
  int*      idx1    = (int*)     take((size_t)NN * CC * 4);
  float*    sim     = (float*)   take((size_t)NN * HW * JC * 4);
  float*    colsc   = (float*)   take((size_t)NN * JC * 4);
  float*    Kmat    = (float*)   take((size_t)NN * CC * HW * 2 * 4);
  float*    maxv    = (float*)   take((size_t)NN * CC * 4);
  float*    rbuf    = (float*)   take((size_t)NN * CC * HW * 4);

  // seeds + per-pixel norms + bf16 transpose
  fill_i32<<<(NN * CC + 255) / 256, 256, 0, stream>>>(cnt, 0, NN * CC);
  k_seed<<<(NN * HW + 255) / 256, 256, 0, stream>>>(norm_cam, label, belong, cnt);
  k_norm<<<(NN * HW + 255) / 256, 256, 0, stream>>>(feature, invn, rowsc);
  k_transpose<<<dim3(HW / 32, CF / 32, NN), dim3(32, 8), 0, stream>>>(feature, invn, featn);

  // deterministic k-means init
  k_scan<<<(NN * CC + 255) / 256, 256, 0, stream>>>(belong, idx0, idx1);
  k_initc<<<NN * JC, 256, 0, stream>>>(feature, idx0, idx1, centers);

  // 10 Lloyd iterations, all 42 centers batched into one WMMA GEMM per image
  for (int iter = 0; iter < 10; ++iter) {
    k_cnorm<<<NN * JC, 256, 0, stream>>>(centers, ctrn, colsc);
    k_gemm<<<(NN * HW / 16) / 8, 256, 0, stream>>>(featn, ctrn, sim);
    fill_f32<<<(NN * JC * CF + 255) / 256, 256, 0, stream>>>(cacc, 0.f, NN * JC * CF);
    fill_u32<<<(NN * JC + 255) / 256, 256, 0, stream>>>(ccnt, 0u, NN * JC);
    k_assign<<<NN * HW / 8, 256, 0, stream>>>(feature, belong, sim, cacc, ccnt);
    k_div<<<(NN * JC * CF + 255) / 256, 256, 0, stream>>>(cacc, ccnt, centers);
  }

  // gate, final cosine GEMM (serves cam AND Sinkhorn sim via row/col rescale)
  k_gate<<<(NN * JC * CF + 255) / 256, 256, 0, stream>>>(cnt, centers);
  k_cnorm<<<NN * JC, 256, 0, stream>>>(centers, ctrn, colsc);
  k_gemm<<<(NN * HW / 16) / 8, 256, 0, stream>>>(featn, ctrn, sim);

  // cam -> out (normalized), Kmat -> Sinkhorn -> T
  k_campost<<<(NN * CC * HW + 255) / 256, 256, 0, stream>>>(sim, cnt, label, rowsc,
                                                            colsc, out1, Kmat);
  k_max<<<NN * CC, 256, 0, stream>>>(out1, maxv);
  k_normout<<<(NN * CC * HW + 255) / 256, 256, 0, stream>>>(out1, maxv);
  k_sinkhorn<<<NN, 1024, 0, stream>>>(Kmat, rbuf, Tout);
}
